// Block_35227321762248
// MI455X (gfx1250) — compile-verified, matrix operand order checked by reading, not run on hardware
//
#include <hip/hip_runtime.h>
#include <hip/hip_bf16.h>

// ---------------------------------------------------------------------------
// Transformer encoder block for MI455X (gfx1250, wave32, WMMA).
// All GEMMs run on v_wmma_f32_16x16x32_bf16; attention is flash-style with
// 64-key tiles. 1/sqrt(d) is folded into the Q projection epilogue.
// ---------------------------------------------------------------------------

typedef __bf16 bf16_t;
typedef __attribute__((ext_vector_type(16))) __bf16 bf16x16;
typedef __attribute__((ext_vector_type(8)))  float  f32x8;

#define DIMC   1024
#define HEADS  16
#define HDIM   64
#define HID    4096
#define NTOK   2048
#define BATCH  2
#define MTOT   (BATCH * NTOK)   // 4096 tokens

// ---------------------------------------------------------------------------
// WMMA helpers
// ---------------------------------------------------------------------------
__device__ inline f32x8 wmma_bf16(bf16x16 a, bf16x16 b, f32x8 c) {
  // (neg_a, A, neg_b, B, c_mod, C, reuse_a, reuse_b)
  return __builtin_amdgcn_wmma_f32_16x16x32_bf16(false, a, false, b,
                                                 (short)0, c, false, false);
}

// Load a 16x32 bf16 fragment (A-style layout; also used for B when the
// source is stored "transposed" i.e. [col][k] row-major).
// ISA 16-bit A layout: lane = m + 16*half;
//   VGPR0..3 : K = half*8 + 0..7
//   VGPR4..7 : K = 16 + half*8 + 0..7
// base points at (row0, k0); ld = row stride in elements (multiple of 8).
__device__ inline bf16x16 load_frag(const bf16_t* base, int ld) {
  int lane = threadIdx.x & 31;
  int m    = lane & 15;
  int half = lane >> 4;
  const bf16_t* p = base + (size_t)m * (size_t)ld + half * 8;
  union { uint4 u[2]; bf16x16 v; } cv;
  cv.u[0] = *reinterpret_cast<const uint4*>(p);
  cv.u[1] = *reinterpret_cast<const uint4*>(p + 16);
  return cv.v;
}

// ---------------------------------------------------------------------------
// Weight prep: fp32 [K][N] -> bf16 transposed [N][K]
// ---------------------------------------------------------------------------
__global__ void wt_transpose_kernel(const float* __restrict__ in,
                                    bf16_t* __restrict__ out, int K, int N) {
  size_t idx = (size_t)blockIdx.x * 256 + threadIdx.x;
  if (idx >= (size_t)K * N) return;
  int k = (int)(idx / (size_t)N);
  int n = (int)(idx % (size_t)N);
  out[(size_t)n * K + k] = (bf16_t)in[idx];
}

// ---------------------------------------------------------------------------
// LayerNorm over DIMC=1024, one block (256 threads) per row. Output bf16.
// ---------------------------------------------------------------------------
__global__ void __launch_bounds__(256) layernorm_kernel(
    const float* __restrict__ x, const float* __restrict__ w,
    const float* __restrict__ b, bf16_t* __restrict__ out) {
  __shared__ float ssum[8];
  __shared__ float ssq[8];
  int row  = blockIdx.x;
  int tid  = threadIdx.x;
  int lane = tid & 31;
  int wv   = tid >> 5;

  const float4* xr = reinterpret_cast<const float4*>(x + (size_t)row * DIMC);
  float4 v = xr[tid];
  float s  = v.x + v.y + v.z + v.w;
  float sq = v.x * v.x + v.y * v.y + v.z * v.z + v.w * v.w;
  #pragma unroll
  for (int mask = 16; mask >= 1; mask >>= 1) {
    s  += __shfl_xor(s, mask, 32);
    sq += __shfl_xor(sq, mask, 32);
  }
  if (lane == 0) { ssum[wv] = s; ssq[wv] = sq; }
  __syncthreads();
  float ts = 0.f, tq = 0.f;
  #pragma unroll
  for (int i = 0; i < 8; ++i) { ts += ssum[i]; tq += ssq[i]; }
  float mu  = ts * (1.0f / DIMC);
  float var = tq * (1.0f / DIMC) - mu * mu;
  float inv = rsqrtf(var + 1e-5f);

  int c = tid * 4;
  const float4 wv4 = reinterpret_cast<const float4*>(w)[tid];
  const float4 bv4 = reinterpret_cast<const float4*>(b)[tid];
  bf16_t* o = out + (size_t)row * DIMC + c;
  o[0] = (bf16_t)((v.x - mu) * inv * wv4.x + bv4.x);
  o[1] = (bf16_t)((v.y - mu) * inv * wv4.y + bv4.y);
  o[2] = (bf16_t)((v.z - mu) * inv * wv4.z + bv4.z);
  o[3] = (bf16_t)((v.w - mu) * inv * wv4.w + bv4.w);
}

// ---------------------------------------------------------------------------
// Generic bf16 WMMA GEMM:  C = (A[M,K] * Bt[N,K]^T + bias) * oscale, epilogue.
// Block = 256 threads (8 waves as 4x2), block tile 128x128,
// wave tile 32x64 (2x4 accumulator tiles), K-step 32.
// ---------------------------------------------------------------------------
enum { EPI_QK = 0, EPI_VT = 1, EPI_RESID_F32 = 2, EPI_GELU_BF16 = 3 };

template <int EPI>
__global__ void __launch_bounds__(256, 4) gemm_kernel(
    const bf16_t* __restrict__ A, const bf16_t* __restrict__ Bt,
    const float* __restrict__ bias, void* __restrict__ outp,
    const float* __restrict__ resid, int M, int N, int K, int Ntok,
    float oscale) {
  int wave = threadIdx.x >> 5;
  int lane = threadIdx.x & 31;
  int wm = wave >> 1;          // 0..3
  int wn = wave & 1;           // 0..1
  int row0 = blockIdx.y * 128 + wm * 32;
  int col0 = blockIdx.x * 128 + wn * 64;

  f32x8 acc[2][4];
  #pragma unroll
  for (int t = 0; t < 2; ++t)
    #pragma unroll
    for (int u = 0; u < 4; ++u) acc[t][u] = (f32x8)(0.0f);

  for (int k0 = 0; k0 < K; k0 += 32) {
    bf16x16 a0 = load_frag(A + (size_t)row0 * K + k0, K);
    bf16x16 a1 = load_frag(A + (size_t)(row0 + 16) * K + k0, K);
    #pragma unroll
    for (int u = 0; u < 4; ++u) {
      bf16x16 bu = load_frag(Bt + (size_t)(col0 + u * 16) * K + k0, K);
      acc[0][u] = wmma_bf16(a0, bu, acc[0][u]);
      acc[1][u] = wmma_bf16(a1, bu, acc[1][u]);
    }
  }

  // Epilogue. C layout: VGPR r, lanes 0-15 -> (M=r, N=lane),
  //                     lanes 16-31 -> (M=r+8, N=lane-16)
  int nloc = lane & 15;
  int moff = (lane >> 4) * 8;
  #pragma unroll
  for (int t = 0; t < 2; ++t) {
    #pragma unroll
    for (int u = 0; u < 4; ++u) {
      #pragma unroll
      for (int r = 0; r < 8; ++r) {
        int rowg = row0 + t * 16 + moff + r;
        int colg = col0 + u * 16 + nloc;
        float val = (acc[t][u][r] + bias[colg]) * oscale;
        if constexpr (EPI == EPI_QK) {
          // -> per-head [b][h][tok][d], bf16
          int bi = rowg / Ntok, tok = rowg % Ntok;
          int h = colg >> 6, d = colg & 63;
          ((bf16_t*)outp)[((((size_t)bi * HEADS + h) * Ntok + tok) << 6) + d] =
              (bf16_t)val;
        } else if constexpr (EPI == EPI_VT) {
          // -> per-head transposed [b][h][d][tok], bf16
          int bi = rowg / Ntok, tok = rowg % Ntok;
          int h = colg >> 6, d = colg & 63;
          ((bf16_t*)outp)[(((size_t)bi * HEADS + h) * HDIM + d) * Ntok + tok] =
              (bf16_t)val;
        } else if constexpr (EPI == EPI_RESID_F32) {
          size_t idx = (size_t)rowg * N + colg;
          ((float*)outp)[idx] = resid[idx] + val;
        } else {  // EPI_GELU_BF16, exact gelu
          float g = 0.5f * val * (1.0f + erff(val * 0.70710678118654752f));
          ((bf16_t*)outp)[(size_t)rowg * N + colg] = (bf16_t)g;
        }
      }
    }
  }
}

// ---------------------------------------------------------------------------
// Flash attention. Grid: (Ntok/128, B*HEADS). Block 256 threads = 8 waves;
// each wave owns a 16-row Q tile and streams keys in chunks of 64.
// q (pre-scaled by 1/sqrt(d)), k: [b][h][tok][64] bf16. vt: [b][h][64][tok].
// out: [b][tok][h*64+d] bf16 (heads re-concatenated for the Wo GEMM).
// ---------------------------------------------------------------------------
#define PLDW 72  // 64 keys + 8 pad

__global__ void __launch_bounds__(256, 4) attn_kernel(
    const bf16_t* __restrict__ q, const bf16_t* __restrict__ k,
    const bf16_t* __restrict__ vt, bf16_t* __restrict__ out, int Ntok) {
  __shared__ bf16_t plds[8][16][PLDW];  // padded 16x64 P tile per wave

  int wave = threadIdx.x >> 5;
  int lane = threadIdx.x & 31;
  int bh = blockIdx.y;
  int q0 = blockIdx.x * 128 + wave * 16;

  const bf16_t* Qh = q  + (size_t)bh * Ntok * HDIM;
  const bf16_t* Kh = k  + (size_t)bh * Ntok * HDIM;
  const bf16_t* Vh = vt + (size_t)bh * HDIM * Ntok;

  bf16x16 qf0 = load_frag(Qh + (size_t)q0 * HDIM + 0, HDIM);
  bf16x16 qf1 = load_frag(Qh + (size_t)q0 * HDIM + 32, HDIM);

  f32x8 oacc[4];
  #pragma unroll
  for (int u = 0; u < 4; ++u) oacc[u] = (f32x8)(0.0f);
  float mrow[8], lrow[8];
  #pragma unroll
  for (int r = 0; r < 8; ++r) { mrow[r] = -1e30f; lrow[r] = 0.0f; }

  int nloc = lane & 15;
  int moff = (lane >> 4) * 8;

  for (int kt = 0; kt < Ntok; kt += 64) {
    // 4 S tiles: 16 q-rows x 64 keys (QK^T, contraction over d=64).
    f32x8 s[4];
    #pragma unroll
    for (int i = 0; i < 4; ++i) {
      const bf16_t* Kt = Kh + (size_t)(kt + i * 16) * HDIM;
      f32x8 si = (f32x8)(0.0f);
      si = wmma_bf16(qf0, load_frag(Kt + 0, HDIM), si);
      si = wmma_bf16(qf1, load_frag(Kt + 32, HDIM), si);
      s[i] = si;
    }

    // Online softmax update; rows live in VGPR r across a 16-lane half.
    #pragma unroll
    for (int r = 0; r < 8; ++r) {
      float a0 = s[0][r], a1 = s[1][r], a2 = s[2][r], a3 = s[3][r];
      float mx = fmaxf(fmaxf(a0, a1), fmaxf(a2, a3));
      mx = fmaxf(mx, __shfl_xor(mx, 1, 32));
      mx = fmaxf(mx, __shfl_xor(mx, 2, 32));
      mx = fmaxf(mx, __shfl_xor(mx, 4, 32));
      mx = fmaxf(mx, __shfl_xor(mx, 8, 32));
      float mnew  = fmaxf(mrow[r], mx);
      float alpha = __expf(mrow[r] - mnew);
      float p0 = __expf(a0 - mnew);
      float p1 = __expf(a1 - mnew);
      float p2 = __expf(a2 - mnew);
      float p3 = __expf(a3 - mnew);
      float rs = (p0 + p1) + (p2 + p3);
      rs += __shfl_xor(rs, 1, 32);
      rs += __shfl_xor(rs, 2, 32);
      rs += __shfl_xor(rs, 4, 32);
      rs += __shfl_xor(rs, 8, 32);
      lrow[r] = lrow[r] * alpha + rs;
      mrow[r] = mnew;
      #pragma unroll
      for (int u = 0; u < 4; ++u) oacc[u][r] *= alpha;
      int rowm = r + moff;
      plds[wave][rowm][nloc]      = (bf16_t)p0;
      plds[wave][rowm][nloc + 16] = (bf16_t)p1;
      plds[wave][rowm][nloc + 32] = (bf16_t)p2;
      plds[wave][rowm][nloc + 48] = (bf16_t)p3;
    }

    // Re-shape P (C-layout -> A-fragment) through LDS, then P @ V.
    #pragma unroll
    for (int ks = 0; ks < 2; ++ks) {
      bf16x16 pf = load_frag(&plds[wave][0][ks * 32], PLDW);
      #pragma unroll
      for (int u = 0; u < 4; ++u) {
        bf16x16 vf =
            load_frag(Vh + (size_t)(u * 16) * Ntok + kt + ks * 32, Ntok);
        oacc[u] = wmma_bf16(pf, vf, oacc[u]);
      }
    }
  }

  // Finalize: divide by l and scatter heads back to [b][tok][DIMC].
  int h  = bh & (HEADS - 1);
  int bi = bh >> 4;
  #pragma unroll
  for (int u = 0; u < 4; ++u) {
    #pragma unroll
    for (int r = 0; r < 8; ++r) {
      int tok = q0 + moff + r;
      int d   = u * 16 + nloc;
      float v = oacc[u][r] / lrow[r];
      out[((size_t)bi * Ntok + tok) * DIMC + h * HDIM + d] = (bf16_t)v;
    }
  }
}

// ---------------------------------------------------------------------------
// Host-side orchestration (graph-capture safe: launches only).
// ---------------------------------------------------------------------------
extern "C" void kernel_launch(void* const* d_in, const int* in_sizes, int n_in,
                              void* d_out, int out_size, void* d_ws, size_t ws_size,
                              hipStream_t stream) {
  (void)in_sizes; (void)n_in; (void)out_size; (void)ws_size;

  const float* x     = (const float*)d_in[0];
  const float* ln1_w = (const float*)d_in[1];
  const float* ln1_b = (const float*)d_in[2];
  const float* ln2_w = (const float*)d_in[3];
  const float* ln2_b = (const float*)d_in[4];
  const float* Wq = (const float*)d_in[5];  const float* bq = (const float*)d_in[6];
  const float* Wk = (const float*)d_in[7];  const float* bk = (const float*)d_in[8];
  const float* Wv = (const float*)d_in[9];  const float* bv = (const float*)d_in[10];
  const float* Wo = (const float*)d_in[11]; const float* bo = (const float*)d_in[12];
  const float* W1 = (const float*)d_in[13]; const float* b1 = (const float*)d_in[14];
  const float* W2 = (const float*)d_in[15]; const float* b2 = (const float*)d_in[16];

  // Workspace layout
  char* ws = (char*)d_ws;
  size_t off = 0;
  auto alloc = [&](size_t bytes) { char* p = ws + off; off += (bytes + 255) & ~size_t(255); return p; };
  bf16_t* Wqt = (bf16_t*)alloc((size_t)DIMC * DIMC * 2);
  bf16_t* Wkt = (bf16_t*)alloc((size_t)DIMC * DIMC * 2);
  bf16_t* Wvt = (bf16_t*)alloc((size_t)DIMC * DIMC * 2);
  bf16_t* Wot = (bf16_t*)alloc((size_t)DIMC * DIMC * 2);
  bf16_t* W1t = (bf16_t*)alloc((size_t)DIMC * HID * 2);
  bf16_t* W2t = (bf16_t*)alloc((size_t)HID * DIMC * 2);
  bf16_t* xn   = (bf16_t*)alloc((size_t)MTOT * DIMC * 2);
  bf16_t* qb   = (bf16_t*)alloc((size_t)MTOT * DIMC * 2);
  bf16_t* kb   = (bf16_t*)alloc((size_t)MTOT * DIMC * 2);
  bf16_t* vtb  = (bf16_t*)alloc((size_t)MTOT * DIMC * 2);
  bf16_t* attn = (bf16_t*)alloc((size_t)MTOT * DIMC * 2);
  float*  x2   = (float*)alloc((size_t)MTOT * DIMC * 4);
  bf16_t* xn2  = (bf16_t*)alloc((size_t)MTOT * DIMC * 2);
  bf16_t* hbuf = (bf16_t*)alloc((size_t)MTOT * HID * 2);

  // 1. Weight prep (fp32 -> bf16 transposed)
  {
    int g1 = (DIMC * DIMC + 255) / 256;
    wt_transpose_kernel<<<g1, 256, 0, stream>>>(Wq, Wqt, DIMC, DIMC);
    wt_transpose_kernel<<<g1, 256, 0, stream>>>(Wk, Wkt, DIMC, DIMC);
    wt_transpose_kernel<<<g1, 256, 0, stream>>>(Wv, Wvt, DIMC, DIMC);
    wt_transpose_kernel<<<g1, 256, 0, stream>>>(Wo, Wot, DIMC, DIMC);
    int g2 = (DIMC * HID + 255) / 256;
    wt_transpose_kernel<<<g2, 256, 0, stream>>>(W1, W1t, DIMC, HID);
    wt_transpose_kernel<<<g2, 256, 0, stream>>>(W2, W2t, HID, DIMC);
  }

  dim3 blk(256);

  // 2. LN1
  layernorm_kernel<<<MTOT, blk, 0, stream>>>(x, ln1_w, ln1_b, xn);

  // 3. Q, K, V projections (WMMA GEMMs). Q is pre-scaled by 1/sqrt(HDIM).
  dim3 gqkv(DIMC / 128, MTOT / 128);
  gemm_kernel<EPI_QK><<<gqkv, blk, 0, stream>>>(xn, Wqt, bq, qb,  nullptr, MTOT, DIMC, DIMC, NTOK, 0.125f);
  gemm_kernel<EPI_QK><<<gqkv, blk, 0, stream>>>(xn, Wkt, bk, kb,  nullptr, MTOT, DIMC, DIMC, NTOK, 1.0f);
  gemm_kernel<EPI_VT><<<gqkv, blk, 0, stream>>>(xn, Wvt, bv, vtb, nullptr, MTOT, DIMC, DIMC, NTOK, 1.0f);

  // 4. Flash attention
  dim3 gatt(NTOK / 128, BATCH * HEADS);
  attn_kernel<<<gatt, blk, 0, stream>>>(qb, kb, vtb, attn, NTOK);

  // 5. Output projection + residual: x2 = x + attn @ Wo + bo
  gemm_kernel<EPI_RESID_F32><<<gqkv, blk, 0, stream>>>(attn, Wot, bo, x2, x, MTOT, DIMC, DIMC, NTOK, 1.0f);

  // 6. LN2
  layernorm_kernel<<<MTOT, blk, 0, stream>>>(x2, ln2_w, ln2_b, xn2);

  // 7. MLP up + GELU: h = gelu(xn2 @ W1 + b1)
  dim3 gup(HID / 128, MTOT / 128);
  gemm_kernel<EPI_GELU_BF16><<<gup, blk, 0, stream>>>(xn2, W1t, b1, hbuf, nullptr, MTOT, HID, DIMC, NTOK, 1.0f);

  // 8. MLP down + residual -> output: out = x2 + h @ W2 + b2
  gemm_kernel<EPI_RESID_F32><<<gqkv, blk, 0, stream>>>(hbuf, W2t, b2, (float*)d_out, x2, MTOT, DIMC, HID, NTOK, 1.0f);
}
